// MultiHeadDotGAT_3968549782097
// MI455X (gfx1250) — compile-verified, matrix operand order checked by reading, not run on hardware
//
#include <hip/hip_runtime.h>
#include <hip/hip_bf16.h>
#include <math.h>

#define N_NODES 20000
#define N_EDGES 640000
#define IN_F 128
#define HID 32
#define HEADS 4
#define OUT_F 128
#define CH (HID * HEADS)          // 128
#define NT (N_NODES / 16)         // 1250 node tiles (exact)
#define TPG 8                     // tiles (waves) per workgroup
#define NGROUPS ((NT + TPG - 1) / TPG)  // 157
#define WSTRIDE 136               // padded LDS row stride in halfs (17*16B rows)

typedef __attribute__((ext_vector_type(16))) _Float16 v16h;
typedef __attribute__((ext_vector_type(8)))  _Float16 v8h;
typedef __attribute__((ext_vector_type(8)))  float    v8f;

// ---------- helpers ----------

__device__ __forceinline__ void atomicMaxF(float* addr, float val) {
  // sign-split trick: correct with init = -inf (0xFF800000)
  if (val >= 0.0f) atomicMax((int*)addr, __float_as_int(val));
  else             atomicMin((unsigned int*)addr, __float_as_uint(val));
}

// Build a v16h WMMA fragment from padded LDS: halves 0..7 at base (K=k0+kb..+7),
// halves 8..15 at base+16 (K=k0+16+kb..+7). Both 16B-aligned -> 2x ds_load_b128.
__device__ __forceinline__ v16h ldsfrag(const _Float16* base) {
  v8h lo = *(const v8h*)(base);
  v8h hi = *(const v8h*)(base + 16);
  return __builtin_shufflevector(lo, hi, 0, 1, 2, 3, 4, 5, 6, 7,
                                 8, 9, 10, 11, 12, 13, 14, 15);
}

// ---------- kernel 0: init scratch (segmax=-inf, denom=0, agg=0) ----------

__global__ void gat_init_kernel(float* __restrict__ segmax,
                                float* __restrict__ denom,
                                float* __restrict__ agg) {
  int i = blockIdx.x * blockDim.x + threadIdx.x;
  if (i < HEADS * N_NODES) {
    segmax[i] = -__builtin_inff();
    denom[i]  = 0.0f;
  }
  if (i < HEADS * N_NODES * HID) agg[i] = 0.0f;
}

// ---------- kernel 1: Q/K projection via WMMA ----------
// Workgroup = 8 waves; grid (NGROUPS, HEADS). Head's qw/kw staged once per
// block into LDS as transposed f16 [d][k] with padded stride; each wave does
// one 16-node tile: 16 WMMAs, B fragments from LDS b128 loads.

__global__ void __launch_bounds__(256)
gat_proj_qk_kernel(const float* __restrict__ x,
                   const float* __restrict__ qw,
                   const float* __restrict__ kw,
                   float* __restrict__ Q,
                   float* __restrict__ K) {
  __shared__ _Float16 ldsQ[HID * WSTRIDE];   // qw[h]^T : [d][k], f16
  __shared__ _Float16 ldsK[HID * WSTRIDE];   // kw[h]^T

  const int tid  = threadIdx.x;
  const int h    = blockIdx.y;
  const float* qwh = qw + (size_t)h * IN_F * HID;
  const float* kwh = kw + (size_t)h * IN_F * HID;

  // stage weights transposed + converted (IN_F*HID = 4096 elems, 16/thread)
#pragma unroll
  for (int t = 0; t < (IN_F * HID) / 256; ++t) {
    const int idx = t * 256 + tid;
    const int d = idx & (HID - 1);
    const int k = idx >> 5;
    ldsQ[d * WSTRIDE + k] = (_Float16)qwh[(size_t)k * HID + d];
    ldsK[d * WSTRIDE + k] = (_Float16)kwh[(size_t)k * HID + d];
  }
  __syncthreads();

  const int wave = tid >> 5;
  const int lane = tid & 31;
  const int tile = blockIdx.x * TPG + wave;
  if (tile >= NT) return;                    // wave-uniform: EXEC stays all-1s

  const int node0 = tile * 16;
  const int m     = lane & 15;
  const int kb    = (lane >> 4) * 8;
  const int n     = lane & 15;

  // A fragments: 16 x-rows, 4 K-steps of 32 (f32 global -> f16 regs)
  const float* xr = x + (size_t)(node0 + m) * IN_F;
  __builtin_prefetch(xr, 0, 1);
  v16h a[4];
#pragma unroll
  for (int ks = 0; ks < 4; ++ks) {
    const int k0 = ks * 32;
#pragma unroll
    for (int j = 0; j < 8; ++j) {
      a[ks][j]     = (_Float16)xr[k0 + kb + j];
      a[ks][8 + j] = (_Float16)xr[k0 + 16 + kb + j];
    }
  }

#pragma unroll
  for (int nt = 0; nt < 2; ++nt) {           // HID=32 -> two 16-wide N tiles
    const int n0 = nt * 16;
    v8f accq = {};
    v8f acck = {};
#pragma unroll
    for (int ks = 0; ks < 4; ++ks) {
      const int k0 = ks * 32;
      const v16h bq = ldsfrag(&ldsQ[(n0 + n) * WSTRIDE + k0 + kb]);
      const v16h bk = ldsfrag(&ldsK[(n0 + n) * WSTRIDE + k0 + kb]);
      accq = __builtin_amdgcn_wmma_f32_16x16x32_f16(false, a[ks], false, bq,
                                                    (short)0, accq, false, false);
      acck = __builtin_amdgcn_wmma_f32_16x16x32_f16(false, a[ks], false, bk,
                                                    (short)0, acck, false, false);
    }
#pragma unroll
    for (int r = 0; r < 8; ++r) {
      const int mm = r + ((lane >> 4) << 3);
      Q[((size_t)h * N_NODES + node0 + mm) * HID + n0 + n] = accq[r];
      K[((size_t)h * N_NODES + node0 + mm) * HID + n0 + n] = acck[r];
    }
  }
}

// ---------- kernel 2: per-edge attention logits + segment max ----------

__global__ void gat_edge_alpha_kernel(const int* __restrict__ row,
                                      const int* __restrict__ col,
                                      const float* __restrict__ Q,
                                      const float* __restrict__ K,
                                      float* __restrict__ alpha,
                                      float* __restrict__ segmax) {
  int idx = blockIdx.x * blockDim.x + threadIdx.x;
  if (idx >= HEADS * N_EDGES) return;
  const int h = idx / N_EDGES;
  const int e = idx - h * N_EDGES;
  const int r = row[e];
  const int c = col[e];
  const float4* qr = (const float4*)(Q + ((size_t)h * N_NODES + r) * HID);
  const float4* kc = (const float4*)(K + ((size_t)h * N_NODES + c) * HID);
  float s = 0.0f;
#pragma unroll
  for (int j = 0; j < 8; ++j) {
    float4 qv = qr[j];
    float4 kv = kc[j];
    s += qv.x * kv.x + qv.y * kv.y + qv.z * kv.z + qv.w * kv.w;
  }
  s *= 0.17677669529663687f;   // 1/sqrt(HID)
  alpha[idx] = s;
  atomicMaxF(&segmax[h * N_NODES + c], s);
}

// ---------- kernel 3: fused softmax-numerator scatter ----------
// agg[h][c][:] += e_a * Q[h][r][:], denom[h][c] += e_a  (normalize later)

__global__ void gat_edge_scatter_kernel(const int* __restrict__ row,
                                        const int* __restrict__ col,
                                        const float* __restrict__ Q,
                                        const float* __restrict__ alpha,
                                        const float* __restrict__ segmax,
                                        float* __restrict__ denom,
                                        float* __restrict__ agg) {
  int idx = blockIdx.x * blockDim.x + threadIdx.x;
  if (idx >= HEADS * N_EDGES) return;
  const int h = idx / N_EDGES;
  const int e = idx - h * N_EDGES;
  const int r = row[e];
  const int c = col[e];
  const float ea = __expf(alpha[idx] - segmax[h * N_NODES + c]);
  atomicAdd(&denom[h * N_NODES + c], ea);
  const float* qr = Q + ((size_t)h * N_NODES + r) * HID;
  float* ag = agg + ((size_t)h * N_NODES + c) * HID;
#pragma unroll
  for (int d = 0; d < HID; ++d) atomicAdd(&ag[d], ea * qr[d]);
}

// ---------- kernel 4: normalize + LayerNorm + SiLU + output GEMM (WMMA) ----
// Workgroup = 8 waves, 8 node tiles. out_w staged once per block as
// transposed f16 in LDS. LN: 2 lanes per row, 64 features in registers,
// wave32 shfl_xor reduction. 32 WMMAs per wave.

__global__ void __launch_bounds__(256)
gat_node_out_kernel(const float* __restrict__ agg,
                    const float* __restrict__ denom,
                    const float* __restrict__ ln_gamma,
                    const float* __restrict__ ln_beta,
                    const float* __restrict__ out_w,
                    const float* __restrict__ out_b,
                    float* __restrict__ out) {
  __shared__ _Float16 ldsW[OUT_F * WSTRIDE];      // out_w^T : [n][k], f16
  __shared__ _Float16 hh[TPG * 16 * WSTRIDE];     // post-LN/SiLU rows, f16
  __shared__ float    ldsG[CH];
  __shared__ float    ldsBt[CH];

  const int tid = threadIdx.x;

  // stage out_w transposed + converted (128*128 = 16384 elems, 64/thread)
#pragma unroll
  for (int t = 0; t < (CH * OUT_F) / 256; ++t) {
    const int idx = t * 256 + tid;
    const int n = idx & (OUT_F - 1);
    const int k = idx >> 7;
    ldsW[n * WSTRIDE + k] = (_Float16)out_w[(size_t)k * OUT_F + n];
  }
  if (tid < CH) {
    ldsG[tid]  = ln_gamma[tid];
    ldsBt[tid] = ln_beta[tid];
  }
  __syncthreads();

  const int wave = tid >> 5;
  const int lane = tid & 31;
  const int tile = blockIdx.x * TPG + wave;
  const bool active = (tile < NT);              // wave-uniform

  if (active) {
    const int node0 = tile * 16;
    // --- normalize by softmax denom + LayerNorm + SiLU, 2 lanes per row ---
    const int r    = lane >> 1;                 // row 0..15
    const int half = lane & 1;                  // feature half (64 each)
    const int node = node0 + r;
    const int h0   = half * 2;                  // heads covered by this lane
    const int h1   = half * 2 + 1;

    float v[64];
    {
      const float den0 = denom[h0 * N_NODES + node];
      const float den1 = denom[h1 * N_NODES + node];
      const float4* a0 = (const float4*)(agg + ((size_t)h0 * N_NODES + node) * HID);
      const float4* a1 = (const float4*)(agg + ((size_t)h1 * N_NODES + node) * HID);
#pragma unroll
      for (int q = 0; q < 8; ++q) {
        const float4 t0 = a0[q];
        v[q * 4 + 0] = t0.x / den0;
        v[q * 4 + 1] = t0.y / den0;
        v[q * 4 + 2] = t0.z / den0;
        v[q * 4 + 3] = t0.w / den0;
      }
#pragma unroll
      for (int q = 0; q < 8; ++q) {
        const float4 t1 = a1[q];
        v[32 + q * 4 + 0] = t1.x / den1;
        v[32 + q * 4 + 1] = t1.y / den1;
        v[32 + q * 4 + 2] = t1.z / den1;
        v[32 + q * 4 + 3] = t1.w / den1;
      }
    }
    float s = 0.0f;
#pragma unroll
    for (int c = 0; c < 64; ++c) s += v[c];
    s += __shfl_xor(s, 1, 32);
    const float mu = s * (1.0f / CH);
    float vs = 0.0f;
#pragma unroll
    for (int c = 0; c < 64; ++c) {
      const float d = v[c] - mu;
      vs += d * d;
    }
    vs += __shfl_xor(vs, 1, 32);
    const float rstd = rsqrtf(vs * (1.0f / CH) + 1e-5f);

    _Float16* hrow = hh + (wave * 16 + r) * WSTRIDE + half * 64;
    const int cbase = half * 64;
#pragma unroll
    for (int c = 0; c < 64; ++c) {
      float hv = (v[c] - mu) * rstd * ldsG[cbase + c] + ldsBt[cbase + c];
      hv = hv / (1.0f + __expf(-hv));           // x * sigmoid(x)
      hrow[c] = (_Float16)hv;
    }
  }
  __syncthreads();                              // uniform barrier

  if (!active) return;

  // --- output GEMM: A = hh tile (16x128), B = ldsW (128x128) ---
  const int node0 = tile * 16;
  const int m  = lane & 15;
  const int kb = (lane >> 4) * 8;
  const int n  = lane & 15;

  v16h a[4];
#pragma unroll
  for (int ks = 0; ks < 4; ++ks)
    a[ks] = ldsfrag(&hh[(wave * 16 + m) * WSTRIDE + ks * 32 + kb]);

#pragma unroll
  for (int nt = 0; nt < 8; ++nt) {              // OUT_F = 128 -> 8 N-tiles
    const int n0 = nt * 16;
    v8f acc = {};
#pragma unroll
    for (int ks = 0; ks < 4; ++ks) {
      const v16h b = ldsfrag(&ldsW[(n0 + n) * WSTRIDE + ks * 32 + kb]);
      acc = __builtin_amdgcn_wmma_f32_16x16x32_f16(false, a[ks], false, b,
                                                   (short)0, acc, false, false);
    }
    const float bias = out_b[n0 + n];
#pragma unroll
    for (int r2 = 0; r2 < 8; ++r2) {
      const int mm = r2 + ((lane >> 4) << 3);
      out[(size_t)(node0 + mm) * OUT_F + n0 + n] = acc[r2] + bias;
    }
  }
}

// ---------- launcher ----------

extern "C" void kernel_launch(void* const* d_in, const int* in_sizes, int n_in,
                              void* d_out, int out_size, void* d_ws, size_t ws_size,
                              hipStream_t stream) {
  const float* x        = (const float*)d_in[0];
  const int*   edge_idx = (const int*)d_in[1];
  const float* qw       = (const float*)d_in[2];
  const float* kw       = (const float*)d_in[3];
  const float* ln_gamma = (const float*)d_in[4];
  const float* ln_beta  = (const float*)d_in[5];
  const float* out_w    = (const float*)d_in[6];
  const float* out_b    = (const float*)d_in[7];
  float* out = (float*)d_out;

  const int* row = edge_idx;                // edge_index[0]
  const int* col = edge_idx + N_EDGES;      // edge_index[1]

  // workspace layout (floats)
  const size_t QK = (size_t)HEADS * N_NODES * HID;  // 2,560,000
  const size_t AE = (size_t)HEADS * N_EDGES;        // 2,560,000
  const size_t HN = (size_t)HEADS * N_NODES;        //    80,000
  float* ws     = (float*)d_ws;
  float* Q      = ws;
  float* K      = Q + QK;
  float* alpha  = K + QK;
  float* segmax = alpha + AE;
  float* denom  = segmax + HN;
  float* agg    = denom + HN;

  // 0) init scratch accumulators
  {
    const int total = HEADS * N_NODES * HID;
    gat_init_kernel<<<(total + 255) / 256, 256, 0, stream>>>(segmax, denom, agg);
  }
  // 1) Q/K projections (WMMA, LDS-staged f16 weights)
  {
    dim3 grid(NGROUPS, HEADS);
    gat_proj_qk_kernel<<<grid, 256, 0, stream>>>(x, qw, kw, Q, K);
  }
  // 2) edge logits + segment max
  {
    const int total = HEADS * N_EDGES;
    gat_edge_alpha_kernel<<<(total + 255) / 256, 256, 0, stream>>>(
        row, col, Q, K, alpha, segmax);
  }
  // 3) fused exp/denominator/aggregate scatter
  {
    const int total = HEADS * N_EDGES;
    gat_edge_scatter_kernel<<<(total + 255) / 256, 256, 0, stream>>>(
        row, col, Q, alpha, segmax, denom, agg);
  }
  // 4) normalize + LN + SiLU + output GEMM (WMMA, LDS-staged f16 weights)
  {
    gat_node_out_kernel<<<NGROUPS, 256, 0, stream>>>(
        agg, denom, ln_gamma, ln_beta, out_w, out_b, out);
  }
}
